// SpeMamba_47425028882458
// MI455X (gfx1250) — matrix operand order repaired and checked by reading, not compiled, not run on hardware
//
#include <hip/hip_runtime.h>
#include <hip/hip_bf16.h>
#include <math.h>

// ---------------- problem constants ----------------
#define B_      4
#define C_      200
#define H_      128
#define W_IMG   128
#define HS      32
#define WSM     32
#define D_MODEL 1024
#define D_STATE 16
#define D_CONV  4
#define D_INNER 2048
#define DT_RANK 64
#define LN_EPS  1e-5f
#define L_      C_
#define BL      (B_ * L_)          // 800 rows = 25 tiles of 32

// ---------------- WMMA types (CDNA5, wave32) ----------------
typedef __attribute__((ext_vector_type(16))) __bf16        v16bf;
typedef __attribute__((ext_vector_type(8)))  float         v8f;
typedef __attribute__((ext_vector_type(8)))  unsigned short v8us;

union Frag16 { v16bf v; v8us h[2]; };

static __device__ __forceinline__ unsigned short f2bf(float f) {
    unsigned int u = __float_as_uint(f);
    unsigned int r = (u + 0x7FFFu + ((u >> 16) & 1u)) >> 16;   // RNE
    return (unsigned short)r;
}
static __device__ __forceinline__ int iclamp(int v, int lo, int hi) {
    return v < lo ? lo : (v > hi ? hi : v);
}

// fragment loaders / MMA group (all indices compile-time -> registers only)
template<int MT>
static __device__ __forceinline__ void loadA(Frag16 (&a)[MT],
                                             const unsigned short* const (&Arow)[MT],
                                             int k0)
{
    #pragma unroll
    for (int s = 0; s < MT; ++s) {
        a[s].h[0] = *(const v8us*)(Arow[s] + k0);
        a[s].h[1] = *(const v8us*)(Arow[s] + k0 + 8);
    }
}
template<int NT>
static __device__ __forceinline__ void loadB(Frag16 (&b)[NT],
                                             const unsigned short* const (&Wrow)[NT],
                                             int k0)
{
    #pragma unroll
    for (int t = 0; t < NT; ++t) {
        b[t].h[0] = *(const v8us*)(Wrow[t] + k0);
        b[t].h[1] = *(const v8us*)(Wrow[t] + k0 + 8);
    }
}
template<int MT, int NT>
static __device__ __forceinline__ void mma(v8f (&acc)[MT][NT],
                                           const Frag16 (&a)[MT],
                                           const Frag16 (&b)[NT])
{
    #pragma unroll
    for (int s = 0; s < MT; ++s)
        #pragma unroll
        for (int t = 0; t < NT; ++t)
            acc[s][t] = __builtin_amdgcn_wmma_f32_16x16x32_bf16(
                false, a[s].v, false, b[t].v, (short)0, acc[s][t], false, false);
}

// ---------------- generic bf16 WMMA GEMM, software-pipelined ----------------
// C[M,N] = A[M,K] * W[N,K]^T   (A,W bf16 row-major; C f32)
// One wave per (16*MT) x (16*NT) tile; statically double-buffered k-loop:
// two 32-wide k-slices in flight (a0/b0, a1/b1), each load group hidden behind
// the 8-WMMA chain of the other slice.  Requires K % 64 == 0 (true here).
// Fragment layouts used here (wave32; contiguous k, same order for A and B):
//   A: lane {m=lane&15, h=lane>>4}: 16 bf16 @ (k0+16h) of row m (contiguous)
//   B: lane {n=lane&15, h=lane>>4}: 16 bf16 @ (k0+16h) of W row n (contiguous)
//   D: VGPR r -> row 8h+r, col lane&15
// EPI: 0 = plain, 1 = +bias, 2 = +bias then softplus
template<int MT, int NT, int EPI>
__global__ __launch_bounds__(32)
void gemm_wmma(const unsigned short* __restrict__ A,
               const unsigned short* __restrict__ W,
               const float* __restrict__ bias,
               float* __restrict__ C,
               int M, int N, int K)
{
    const int lane  = threadIdx.x & 31;
    const int m16   = lane & 15;
    const int hh    = lane >> 4;
    const int tileM = blockIdx.y * (16 * MT);
    const int tileN = blockIdx.x * (16 * NT);

    const unsigned short* Arow[MT];
    #pragma unroll
    for (int s = 0; s < MT; ++s)
        Arow[s] = A + (size_t)(tileM + s * 16 + m16) * K + 16 * hh;
    const unsigned short* Wrow[NT];
    #pragma unroll
    for (int t = 0; t < NT; ++t)
        Wrow[t] = W + (size_t)(tileN + t * 16 + m16) * K + 16 * hh;

    v8f acc[MT][NT];
    #pragma unroll
    for (int s = 0; s < MT; ++s)
        #pragma unroll
        for (int t = 0; t < NT; ++t)
            #pragma unroll
            for (int i = 0; i < 8; ++i) acc[s][t][i] = 0.0f;

    Frag16 a0[MT], b0[NT], a1[MT], b1[NT];
    loadA(a0, Arow, 0);   loadB(b0, Wrow, 0);
    loadA(a1, Arow, 32);  loadB(b1, Wrow, 32);

    for (int k = 64; k < K; k += 64) {
        mma(acc, a0, b0);
        loadA(a0, Arow, k);      loadB(b0, Wrow, k);
        mma(acc, a1, b1);
        loadA(a1, Arow, k + 32); loadB(b1, Wrow, k + 32);
    }
    mma(acc, a0, b0);
    mma(acc, a1, b1);

    #pragma unroll
    for (int s = 0; s < MT; ++s)
        #pragma unroll
        for (int t = 0; t < NT; ++t) {
            const int col = tileN + t * 16 + m16;
            #pragma unroll
            for (int r = 0; r < 8; ++r) {
                const int row = tileM + s * 16 + 8 * hh + r;
                float v = acc[s][t][r];
                if (EPI >= 1) v += bias[col];
                if (EPI == 2) v = (v > 20.0f) ? v : log1pf(__expf(v));
                C[(size_t)row * N + col] = v;
            }
        }
}

// ---------------- elementwise kernels ----------------
__global__ void cvt_f32_bf16_kernel(const float* __restrict__ src,
                                    unsigned short* __restrict__ dst, int n)
{
    int gid = blockIdx.x * blockDim.x + threadIdx.x;
    if (gid < n) dst[gid] = f2bf(src[gid]);
}

// 4x4 average pool: x[B,C,128,128] -> xf_bf16[B*C,1024]
__global__ void pool_kernel(const float* __restrict__ x,
                            unsigned short* __restrict__ xf)
{
    int gid = blockIdx.x * blockDim.x + threadIdx.x;
    if (gid >= B_ * C_ * D_MODEL) return;
    int p  = gid & (D_MODEL - 1);
    int bc = gid >> 10;
    int ph = p >> 5, pw = p & 31;
    const float* base = x + (size_t)bc * H_ * W_IMG + (ph * 4) * W_IMG + pw * 4;
    float s = 0.0f;
    #pragma unroll
    for (int dy = 0; dy < 4; ++dy)
        #pragma unroll
        for (int dx = 0; dx < 4; ++dx) s += base[dy * W_IMG + dx];
    xf[gid] = f2bf(s * 0.0625f);
}

// depthwise causal conv along L + bias + SiLU; input xs = xz[..., 0:D_INNER]
__global__ void conv_silu_kernel(const float* __restrict__ xz,
                                 const float* __restrict__ conv_w,
                                 const float* __restrict__ conv_b,
                                 float* __restrict__ u,
                                 unsigned short* __restrict__ u_bf)
{
    int gid = blockIdx.x * blockDim.x + threadIdx.x;
    if (gid >= BL * D_INNER) return;
    int d  = gid & (D_INNER - 1);
    int bl = gid >> 11;
    int l  = bl % L_;
    int b  = bl / L_;
    float acc = conv_b[d];
    #pragma unroll
    for (int j = 0; j < D_CONV; ++j) {
        int ls = l - (D_CONV - 1) + j;
        if (ls >= 0)
            acc += conv_w[d * D_CONV + j] *
                   xz[((size_t)(b * L_ + ls)) * (2 * D_INNER) + d];
    }
    float s = acc / (1.0f + __expf(-acc));
    u[gid]    = s;
    u_bf[gid] = f2bf(s);
}

// copy dt columns (first 64 of dbc's 96) to packed bf16 [BL,64]
__global__ void extract_dt_kernel(const float* __restrict__ dbc,
                                  unsigned short* __restrict__ dt)
{
    int gid = blockIdx.x * blockDim.x + threadIdx.x;
    if (gid >= BL * DT_RANK) return;
    int r  = gid & 63;
    int bl = gid >> 6;
    dt[gid] = f2bf(dbc[(size_t)bl * 96 + r]);
}

// selective scan: 8192 channels, 16 f32 states each in registers; L=200 serial
__global__ __launch_bounds__(256)
void scan_kernel(const float* __restrict__ delta, const float* __restrict__ u,
                 const float* __restrict__ xz,    const float* __restrict__ dbc,
                 const float* __restrict__ A_log, const float* __restrict__ Dp,
                 unsigned short* __restrict__ yg)
{
    __shared__ float sB[D_STATE], sC[D_STATE];
    const int tid = threadIdx.x;
    const int b   = blockIdx.x >> 3;
    const int d   = ((blockIdx.x & 7) << 8) + tid;

    float A[D_STATE], h[D_STATE];
    #pragma unroll
    for (int s = 0; s < D_STATE; ++s) {
        A[s] = -__expf(A_log[d * D_STATE + s]);
        h[s] = 0.0f;
    }
    const float Dd = Dp[d];

    for (int l = 0; l < L_; ++l) {
        const size_t base = (size_t)(b * L_ + l);
        if (tid < 2 * D_STATE) {
            float v = dbc[base * 96 + 64 + tid];
            if (tid < D_STATE) sB[tid] = v;
            else               sC[tid - D_STATE] = v;
        }
        __syncthreads();
        const float dl = delta[base * D_INNER + d];
        const float ul = u[base * D_INNER + d];
        const float z  = xz[base * (2 * D_INNER) + D_INNER + d];
        const float du = dl * ul;
        float y = 0.0f;
        #pragma unroll
        for (int s = 0; s < D_STATE; ++s) {
            float dA = __expf(dl * A[s]);
            h[s] = dA * h[s] + du * sB[s];
            y += h[s] * sC[s];
        }
        y += Dd * ul;
        const float sz = z / (1.0f + __expf(-z));
        yg[base * D_INNER + d] = f2bf(y * sz);
        __syncthreads();
    }
}

// LayerNorm over d_model=1024 + SiLU, in place (one block per row)
__global__ __launch_bounds__(256)
void ln_silu_kernel(float* __restrict__ xp,
                    const float* __restrict__ g, const float* __restrict__ bb)
{
    __shared__ float red[256], red2[256];
    const int row = blockIdx.x;
    const int tid = threadIdx.x;
    float v[4], s = 0.0f, s2 = 0.0f;
    #pragma unroll
    for (int i = 0; i < 4; ++i) {
        v[i] = xp[(size_t)row * D_MODEL + tid + i * 256];
        s += v[i]; s2 += v[i] * v[i];
    }
    red[tid] = s; red2[tid] = s2;
    __syncthreads();
    for (int off = 128; off > 0; off >>= 1) {
        if (tid < off) { red[tid] += red[tid + off]; red2[tid] += red2[tid + off]; }
        __syncthreads();
    }
    const float mu   = red[0] * (1.0f / D_MODEL);
    const float var  = red2[0] * (1.0f / D_MODEL) - mu * mu;
    const float rstd = rsqrtf(var + LN_EPS);
    #pragma unroll
    for (int i = 0; i < 4; ++i) {
        const int dc = tid + i * 256;
        float t = (v[i] - mu) * rstd * g[dc] + bb[dc];
        xp[(size_t)row * D_MODEL + dc] = t / (1.0f + __expf(-t));
    }
}

// bilinear x4 upsample (half-pixel centers) + residual add
__global__ void upsample_add_kernel(const float* __restrict__ x,
                                    const float* __restrict__ xp,
                                    float* __restrict__ out)
{
    size_t gid = (size_t)blockIdx.x * blockDim.x + threadIdx.x;
    if (gid >= (size_t)B_ * C_ * H_ * W_IMG) return;
    const int    j  = gid & 127;
    const int    i  = (gid >> 7) & 127;
    const size_t bc = gid >> 14;
    const float sy = (i + 0.5f) * 0.25f - 0.5f;
    const float sx = (j + 0.5f) * 0.25f - 0.5f;
    const float fy = sy - floorf(sy);
    const float fx = sx - floorf(sx);
    int y0 = (int)floorf(sy), x0 = (int)floorf(sx);
    const int y1 = iclamp(y0 + 1, 0, HS - 1);
    const int x1 = iclamp(x0 + 1, 0, WSM - 1);
    y0 = iclamp(y0, 0, HS - 1);
    x0 = iclamp(x0, 0, WSM - 1);
    const float* p = xp + bc * (HS * WSM);
    const float v00 = p[y0 * WSM + x0], v01 = p[y0 * WSM + x1];
    const float v10 = p[y1 * WSM + x0], v11 = p[y1 * WSM + x1];
    const float v = v00 * (1 - fy) * (1 - fx) + v01 * (1 - fy) * fx +
                    v10 * fy * (1 - fx)       + v11 * fy * fx;
    out[gid] = x[gid] + v;
}

// ---------------- host ----------------
extern "C" void kernel_launch(void* const* d_in, const int* in_sizes, int n_in,
                              void* d_out, int out_size, void* d_ws, size_t ws_size,
                              hipStream_t stream)
{
    (void)in_sizes; (void)n_in; (void)out_size; (void)ws_size;
    const float* x      = (const float*)d_in[0];
    const float* W_in   = (const float*)d_in[1];
    const float* conv_w = (const float*)d_in[2];
    const float* conv_b = (const float*)d_in[3];
    const float* W_xproj= (const float*)d_in[4];
    const float* W_dt   = (const float*)d_in[5];
    const float* b_dt   = (const float*)d_in[6];
    const float* A_log  = (const float*)d_in[7];
    const float* Dp     = (const float*)d_in[8];
    const float* W_out  = (const float*)d_in[9];
    const float* W_proj = (const float*)d_in[10];
    const float* b_proj = (const float*)d_in[11];
    const float* ln_g   = (const float*)d_in[12];
    const float* ln_b   = (const float*)d_in[13];
    float* out = (float*)d_out;

    char*  ws  = (char*)d_ws;
    size_t off = 0;
    auto alloc = [&](size_t bytes) -> char* {
        char* p = ws + off;
        off += (bytes + 255) & ~(size_t)255;
        return p;
    };
    unsigned short* wInB    = (unsigned short*)alloc((size_t)2 * D_INNER * D_MODEL * 2);  // [4096,1024]
    unsigned short* wXprojB = (unsigned short*)alloc((size_t)96 * D_INNER * 2);           // [96,2048]
    unsigned short* wDtB    = (unsigned short*)alloc((size_t)D_INNER * DT_RANK * 2);      // [2048,64]
    unsigned short* wOutB   = (unsigned short*)alloc((size_t)D_MODEL * D_INNER * 2);      // [1024,2048]
    unsigned short* wProjB  = (unsigned short*)alloc((size_t)D_MODEL * D_MODEL * 2);      // [1024,1024]
    unsigned short* xfB     = (unsigned short*)alloc((size_t)BL * D_MODEL * 2);
    float*          xz      = (float*)alloc((size_t)BL * 2 * D_INNER * 4);
    float*          u       = (float*)alloc((size_t)BL * D_INNER * 4);
    unsigned short* uB      = (unsigned short*)alloc((size_t)BL * D_INNER * 2);
    float*          dbc     = (float*)alloc((size_t)BL * 96 * 4);
    unsigned short* dtB     = (unsigned short*)alloc((size_t)BL * DT_RANK * 2);
    float*          delta   = (float*)alloc((size_t)BL * D_INNER * 4);
    unsigned short* ygB     = (unsigned short*)alloc((size_t)BL * D_INNER * 2);
    float*          xm      = (float*)alloc((size_t)BL * D_MODEL * 4);
    unsigned short* xmB     = (unsigned short*)alloc((size_t)BL * D_MODEL * 2);
    float*          xp      = (float*)alloc((size_t)BL * D_MODEL * 4);

    const int T = 256;
    // weights -> bf16 (L2-resident afterwards; 192 MB L2 holds everything)
    cvt_f32_bf16_kernel<<<(2*D_INNER*D_MODEL + T-1)/T, T, 0, stream>>>(W_in,    wInB,    2*D_INNER*D_MODEL);
    cvt_f32_bf16_kernel<<<(96*D_INNER       + T-1)/T, T, 0, stream>>>(W_xproj, wXprojB, 96*D_INNER);
    cvt_f32_bf16_kernel<<<(D_INNER*DT_RANK  + T-1)/T, T, 0, stream>>>(W_dt,    wDtB,    D_INNER*DT_RANK);
    cvt_f32_bf16_kernel<<<(D_MODEL*D_INNER  + T-1)/T, T, 0, stream>>>(W_out,   wOutB,   D_MODEL*D_INNER);
    cvt_f32_bf16_kernel<<<(D_MODEL*D_MODEL  + T-1)/T, T, 0, stream>>>(W_proj,  wProjB,  D_MODEL*D_MODEL);

    // 1) pool -> xf bf16
    pool_kernel<<<(BL*D_MODEL + T-1)/T, T, 0, stream>>>(x, xfB);
    // 2) xz = xf @ W_in^T   [800,4096] K=1024
    gemm_wmma<2,4,0><<<dim3((2*D_INNER)/64, BL/32), 32, 0, stream>>>(xfB, wInB, nullptr, xz, BL, 2*D_INNER, D_MODEL);
    // 3) depthwise conv + SiLU -> u (f32 + bf16)
    conv_silu_kernel<<<(BL*D_INNER + T-1)/T, T, 0, stream>>>(xz, conv_w, conv_b, u, uB);
    // 4) dbc = u @ W_xproj^T  [800,96] K=2048
    gemm_wmma<2,2,0><<<dim3(96/32, BL/32), 32, 0, stream>>>(uB, wXprojB, nullptr, dbc, BL, 96, D_INNER);
    // 5) dt -> bf16
    extract_dt_kernel<<<(BL*DT_RANK + T-1)/T, T, 0, stream>>>(dbc, dtB);
    // 6) delta = softplus(dt @ W_dt^T + b_dt)  [800,2048] K=64
    gemm_wmma<2,4,2><<<dim3(D_INNER/64, BL/32), 32, 0, stream>>>(dtB, wDtB, b_dt, delta, BL, D_INNER, DT_RANK);
    // 7) selective scan + skip + gate -> yg bf16
    scan_kernel<<<(B_*D_INNER)/T, T, 0, stream>>>(delta, u, xz, dbc, A_log, Dp, ygB);
    // 8) xm = yg @ W_out^T  [800,1024] K=2048
    gemm_wmma<2,4,0><<<dim3(D_MODEL/64, BL/32), 32, 0, stream>>>(ygB, wOutB, nullptr, xm, BL, D_MODEL, D_INNER);
    cvt_f32_bf16_kernel<<<(BL*D_MODEL + T-1)/T, T, 0, stream>>>(xm, xmB, BL*D_MODEL);
    // 9) xp = xm @ W_proj^T + b_proj  [800,1024] K=1024
    gemm_wmma<2,4,1><<<dim3(D_MODEL/64, BL/32), 32, 0, stream>>>(xmB, wProjB, b_proj, xp, BL, D_MODEL, D_MODEL);
    // 10) LayerNorm + SiLU (in place)
    ln_silu_kernel<<<BL, 256, 0, stream>>>(xp, ln_g, ln_b);
    // 11) bilinear x4 upsample + residual
    upsample_add_kernel<<<(int)(((size_t)B_*C_*H_*W_IMG + T-1)/T), T, 0, stream>>>(x, xp, out);
}
